// CGCNN_53549652246908
// MI455X (gfx1250) — matrix operand order repaired and checked by reading, not compiled, not run on hardware
//
#include <hip/hip_runtime.h>
#include <stdint.h>

#define N_SITES    100000
#define N_EDGES    1600000
#define N_GRAPHS   512
#define SITE_PROPS 92
#define EMB        64
#define KPAD_SE    96
#define KCAT       192
#define HH1        128
#define HH2        64

typedef __attribute__((ext_vector_type(16))) _Float16 v16h;
typedef __attribute__((ext_vector_type(8)))  float    v8f;
typedef __attribute__((ext_vector_type(2)))  __fp16   fp16x2;

// Fragment: 16 f16 = 8 dwords = 2 x 128-bit
union Frag {
  v16h h;
  uint32_t u[8];
  uint4 q[2];
};

// A-matrix (16x32 f16) per-lane k offset for dword j (two contiguous k's)
__device__ __forceinline__ int kA(int j, int half) {
  return ((j >> 2) << 4) + (half << 3) + ((j & 3) << 1);
}

// two f32 -> one packed f16 dword in a single v_cvt_pk_rtz_f16_f32
__device__ __forceinline__ uint32_t pack2h(float a, float b) {
  union { fp16x2 h; uint32_t u; } x;
  x.h = __builtin_amdgcn_cvt_pkrtz(a, b);
  return x.u;
}

// B-fragment for one 32-k step: lane covers halfs [8*half, 8*half+8) and
// [16+8*half, 16+8*half+8) of the lane's weight row -> two b128 loads.
__device__ __forceinline__ v16h load_bfrag(const _Float16* wrow_k, int half) {
  Frag b;
  const uint4* p = (const uint4*)(wrow_k + (half << 3));
  b.q[0] = p[0];
  b.q[1] = p[2];
  return b.h;
}

// ---------------------------------------------------------------- weight prep
__global__ void prep_weights(const float* W_se, const float* W_be,
                             const float* Ws1, const float* Wo1,
                             const float* Ws2, const float* Wo2,
                             _Float16* t_se, _Float16* t_be,
                             _Float16* t_s1, _Float16* t_o1,
                             _Float16* t_s2, _Float16* t_o2) {
  int i = blockIdx.x * blockDim.x + threadIdx.x;
  if (i < EMB * KPAD_SE) {            // [n][k] k-contig, zero-padded to 96
    int n = i / KPAD_SE, k = i % KPAD_SE;
    t_se[i] = (_Float16)((k < SITE_PROPS) ? W_se[k * EMB + n] : 0.0f);
  }
  if (i < EMB * EMB) {
    int n = i / EMB, k = i % EMB;
    t_be[i] = (_Float16)W_be[k * EMB + n];
  }
  if (i < EMB * KCAT) {
    int n = i / KCAT, k = i % KCAT;
    t_s1[i] = (_Float16)Ws1[k * EMB + n];
    t_o1[i] = (_Float16)Wo1[k * EMB + n];
    t_s2[i] = (_Float16)Ws2[k * EMB + n];
    t_o2[i] = (_Float16)Wo2[k * EMB + n];
  }
}

__global__ void zero_f32(float* p, long n) {
  long i = (long)blockIdx.x * blockDim.x + threadIdx.x;
  long stride = (long)gridDim.x * blockDim.x;
  for (; i < n; i += stride) p[i] = 0.0f;
}

// ------------------------------------------------------------- site embedding
__global__ void site_embed(const float* __restrict__ sites,
                           const _Float16* __restrict__ t_se,
                           const float* __restrict__ b_se,
                           float* __restrict__ s, _Float16* __restrict__ sh) {
  const int lane = threadIdx.x & 31;
  const int wave = threadIdx.x >> 5;
  const int nt   = wave & 3;
  const int sub  = wave >> 2;
  const int half = lane >> 4;
  const int mn   = lane & 15;
  const int n    = nt * 16 + mn;

  v16h B[3];
  {
    const _Float16* wp = t_se + (long)n * KPAD_SE;
#pragma unroll
    for (int kk = 0; kk < 3; ++kk) B[kk] = load_bfrag(wp + kk * 32, half);
  }
  const float bias = b_se[n];

  const int PAIRS = (N_SITES / 16) / 2;  // 3125
  for (int p = blockIdx.x; p < PAIRS; p += gridDim.x) {
    const long tile = (long)p * 2 + sub;
    const float* ap = sites + (tile * 16 + mn) * SITE_PROPS;
    v8f c = {};
#pragma unroll
    for (int kk = 0; kk < 3; ++kk) {
      Frag a;
#pragma unroll
      for (int j = 0; j < 8; ++j) {
        int k = kk * 32 + kA(j, half);
        float f0 = (k     < SITE_PROPS) ? ap[k]     : 0.0f;
        float f1 = (k + 1 < SITE_PROPS) ? ap[k + 1] : 0.0f;
        a.u[j] = pack2h(f0, f1);
      }
      c = __builtin_amdgcn_wmma_f32_16x16x32_f16(false, a.h, false, B[kk],
                                                 (short)0, c, false, false);
    }
#pragma unroll
    for (int r = 0; r < 8; ++r) {
      long row = tile * 16 + r + (half << 3);
      float v = c[r] + bias;
      s [row * EMB + n] = v;
      sh[row * EMB + n] = (_Float16)v;
    }
  }
}

// ------------------------------------------------------------- bond embedding
__global__ void bond_embed(const float* __restrict__ bonds,
                           const _Float16* __restrict__ t_be,
                           const float* __restrict__ b_be,
                           _Float16* __restrict__ beh) {
  const int lane = threadIdx.x & 31;
  const int wave = threadIdx.x >> 5;
  const int nt   = wave & 3;
  const int sub  = wave >> 2;
  const int half = lane >> 4;
  const int mn   = lane & 15;
  const int n    = nt * 16 + mn;

  v16h B[2];
  {
    const _Float16* wp = t_be + (long)n * EMB;
#pragma unroll
    for (int kk = 0; kk < 2; ++kk) B[kk] = load_bfrag(wp + kk * 32, half);
  }
  const float bias = b_be[n];

  const int PAIRS = (N_EDGES / 16) / 2;  // 50000
  for (int p = blockIdx.x; p < PAIRS; p += gridDim.x) {
    const long tile = (long)p * 2 + sub;
    const float d = bonds[tile * 16 + mn];
    v8f c = {};
#pragma unroll
    for (int kk = 0; kk < 2; ++kk) {
      Frag a;
#pragma unroll
      for (int j = 0; j < 8; ++j) {
        int k = kk * 32 + kA(j, half);
        float d0 = d - (float)k * 0.125f;
        float d1 = d0 - 0.125f;
        a.u[j] = pack2h(__expf(-d0 * d0 * 64.0f), __expf(-d1 * d1 * 64.0f));
      }
      c = __builtin_amdgcn_wmma_f32_16x16x32_f16(false, a.h, false, B[kk],
                                                 (short)0, c, false, false);
    }
#pragma unroll
    for (int r = 0; r < 8; ++r) {
      long row = tile * 16 + r + (half << 3);
      beh[row * EMB + n] = (_Float16)(c[r] + bias);
    }
  }
}

// ------------------------------------------------------------------ conv layer
// per edge: v=[s[i1], s[i2], be]  g=sigmoid(v@Ws+bs)*relu(v@Wo+bo)
// agg[i1] += g ; 16 edges x 16 outputs per wave; A frags = 2x b128 per k-step
__global__ void conv_layer(const _Float16* __restrict__ sh,
                           const _Float16* __restrict__ beh,
                           const int* __restrict__ i1g,
                           const int* __restrict__ i2g,
                           const _Float16* __restrict__ t_sig,
                           const float* __restrict__ b_sig,
                           const _Float16* __restrict__ t_sof,
                           const float* __restrict__ b_sof,
                           float* __restrict__ agg) {
  const int lane = threadIdx.x & 31;
  const int wave = threadIdx.x >> 5;
  const int nt   = wave & 3;
  const int sub  = wave >> 2;
  const int half = lane >> 4;
  const int mn   = lane & 15;
  const int n    = nt * 16 + mn;

  v16h Bs[6], Bo[6];
  {
    const _Float16* ws = t_sig + (long)n * KCAT;
    const _Float16* wo = t_sof + (long)n * KCAT;
#pragma unroll
    for (int kk = 0; kk < 6; ++kk) {
      Bs[kk] = load_bfrag(ws + kk * 32, half);
      Bo[kk] = load_bfrag(wo + kk * 32, half);
    }
  }
  const float bs_ = b_sig[n];
  const float bo_ = b_sof[n];

  const int PAIRS = (N_EDGES / 16) / 2;  // 50000
  for (int p = blockIdx.x; p < PAIRS; p += gridDim.x) {
    const long base = ((long)p * 2 + sub) * 16;
    const int s1 = i1g[base + mn];
    const int s2 = i2g[base + mn];
    const _Float16* p1 = sh  + (long)s1 * EMB + (half << 3);
    const _Float16* p2 = sh  + (long)s2 * EMB + (half << 3);
    const _Float16* pb = beh + (base + mn) * EMB + (half << 3);

    v8f cs = {}, co = {};
#pragma unroll
    for (int kk = 0; kk < 6; ++kk) {
      // segment base (uniform per kk after unroll), offset within segment
      const _Float16* seg = (kk < 2) ? p1 : (kk < 4) ? p2 : pb;
      const int kbase = (kk & 1) << 5;  // 0 or 32 halfs within the 64-wide segment
      Frag a;
      a.q[0] = *(const uint4*)(seg + kbase);        // halfs [kbase+8h, +8)
      a.q[1] = *(const uint4*)(seg + kbase + 16);   // halfs [kbase+16+8h, +8)
      cs = __builtin_amdgcn_wmma_f32_16x16x32_f16(false, a.h, false, Bs[kk],
                                                  (short)0, cs, false, false);
      co = __builtin_amdgcn_wmma_f32_16x16x32_f16(false, a.h, false, Bo[kk],
                                                  (short)0, co, false, false);
    }
#pragma unroll
    for (int r = 0; r < 8; ++r) {
      long e = base + r + (half << 3);
      float xs = cs[r] + bs_;
      float xo = co[r] + bo_;
      float g = (1.0f / (1.0f + __expf(-xs))) * fmaxf(xo, 0.0f);
      int dst = i1g[e];
      atomicAdd(agg + (long)dst * EMB + n, g);
    }
  }
}

// -------------------------------------------------------- residual + f16 copy
__global__ void residual_add(float* __restrict__ s, const float* __restrict__ agg,
                             _Float16* __restrict__ sh, long nElem) {
  long i = (long)blockIdx.x * blockDim.x + threadIdx.x;
  long stride = (long)gridDim.x * blockDim.x;
  for (; i < nElem; i += stride) {
    float v = s[i] + agg[i];
    s[i] = v;
    sh[i] = (_Float16)v;
  }
}

// -------------------------------------------------------------- graph pooling
__global__ void pool_graphs(const float* __restrict__ s,
                            const int* __restrict__ g2s,
                            float* __restrict__ sums, float* __restrict__ cnts) {
  long i = (long)blockIdx.x * blockDim.x + threadIdx.x;
  long stride = (long)gridDim.x * blockDim.x;
  const long total = (long)N_SITES * EMB;
  for (; i < total; i += stride) {
    int site = (int)(i >> 6);
    int n = (int)(i & 63);
    int g = g2s[site];
    atomicAdd(&sums[(long)g * EMB + n], s[i]);
    if (n == 0) atomicAdd(&cnts[g], 1.0f);
  }
}

// ------------------------------------------------------------------- head MLP
__global__ void head_mlp(const float* __restrict__ sums, const float* __restrict__ cnts,
                         const float* __restrict__ W1, const float* __restrict__ b1,
                         const float* __restrict__ W2, const float* __restrict__ b2,
                         const float* __restrict__ W3, const float* __restrict__ b3,
                         float* __restrict__ out) {
  __shared__ float vec[EMB];
  __shared__ float h1[HH1];
  __shared__ float h2[HH2];
  const int g = blockIdx.x;
  const int t = threadIdx.x;
  const float c = fmaxf(cnts[g], 1.0f);
  if (t < EMB) vec[t] = sums[(long)g * EMB + t] / c;
  __syncthreads();
  {
    float a = b1[t];
    for (int k = 0; k < EMB; ++k) a += vec[k] * W1[k * HH1 + t];
    h1[t] = fmaxf(a, 0.0f);
  }
  __syncthreads();
  if (t < HH2) {
    float a = b2[t];
    for (int k = 0; k < HH1; ++k) a += h1[k] * W2[k * HH2 + t];
    h2[t] = fmaxf(a, 0.0f);
  }
  __syncthreads();
  if (t == 0) {
    float a = b3[0];
    for (int k = 0; k < HH2; ++k) a += h2[k] * W3[k];
    out[g] = a;
  }
}

// ------------------------------------------------------------------- launcher
extern "C" void kernel_launch(void* const* d_in, const int* in_sizes, int n_in,
                              void* d_out, int out_size, void* d_ws, size_t ws_size,
                              hipStream_t stream) {
  (void)in_sizes; (void)n_in; (void)out_size; (void)ws_size;

  const float* sites  = (const float*)d_in[0];
  const float* bonds  = (const float*)d_in[1];
  const int*   i1     = (const int*)d_in[2];
  const int*   i2     = (const int*)d_in[3];
  const int*   g2s    = (const int*)d_in[4];
  const float* W_se   = (const float*)d_in[5];
  const float* b_se   = (const float*)d_in[6];
  const float* W_be   = (const float*)d_in[7];
  const float* b_be   = (const float*)d_in[8];
  const float* W_sig1 = (const float*)d_in[9];
  const float* b_sig1 = (const float*)d_in[10];
  const float* W_sof1 = (const float*)d_in[11];
  const float* b_sof1 = (const float*)d_in[12];
  const float* W_sig2 = (const float*)d_in[13];
  const float* b_sig2 = (const float*)d_in[14];
  const float* W_sof2 = (const float*)d_in[15];
  const float* b_sof2 = (const float*)d_in[16];
  const float* W1     = (const float*)d_in[17];
  const float* b1     = (const float*)d_in[18];
  const float* W2     = (const float*)d_in[19];
  const float* b2     = (const float*)d_in[20];
  const float* W3     = (const float*)d_in[21];
  const float* b3     = (const float*)d_in[22];
  float* out = (float*)d_out;

  // ---- carve workspace
  char* w = (char*)d_ws;
  size_t off = 0;
  auto carve = [&](size_t bytes) -> void* {
    void* p = w + off;
    off = (off + bytes + 255) & ~(size_t)255;
    return p;
  };
  float*    s_f32 = (float*)   carve((size_t)N_SITES * EMB * 4);
  _Float16* s_h   = (_Float16*)carve((size_t)N_SITES * EMB * 2);
  float*    agg   = (float*)   carve((size_t)N_SITES * EMB * 4);
  _Float16* beh   = (_Float16*)carve((size_t)N_EDGES * EMB * 2);
  _Float16* t_se  = (_Float16*)carve((size_t)EMB * KPAD_SE * 2);
  _Float16* t_be  = (_Float16*)carve((size_t)EMB * EMB * 2);
  _Float16* t_s1  = (_Float16*)carve((size_t)EMB * KCAT * 2);
  _Float16* t_o1  = (_Float16*)carve((size_t)EMB * KCAT * 2);
  _Float16* t_s2  = (_Float16*)carve((size_t)EMB * KCAT * 2);
  _Float16* t_o2  = (_Float16*)carve((size_t)EMB * KCAT * 2);
  float*    sums  = (float*)   carve((size_t)N_GRAPHS * EMB * 4);
  float*    cnts  = (float*)   carve((size_t)N_GRAPHS * 4);

  const long sElem = (long)N_SITES * EMB;

  // 1) weights -> transposed f16
  prep_weights<<<(EMB * KCAT + 255) / 256, 256, 0, stream>>>(
      W_se, W_be, W_sig1, W_sof1, W_sig2, W_sof2,
      t_se, t_be, t_s1, t_o1, t_s2, t_o2);

  // 2) site + bond embeddings (WMMA)
  site_embed<<<1536, 256, 0, stream>>>(sites, t_se, b_se, s_f32, s_h);
  bond_embed<<<4096, 256, 0, stream>>>(bonds, t_be, b_be, beh);

  // 3) conv layer 1
  zero_f32<<<2048, 256, 0, stream>>>(agg, sElem);
  conv_layer<<<4096, 256, 0, stream>>>(s_h, beh, i1, i2, t_s1, b_sig1, t_o1, b_sof1, agg);
  residual_add<<<2048, 256, 0, stream>>>(s_f32, agg, s_h, sElem);

  // 4) conv layer 2
  zero_f32<<<2048, 256, 0, stream>>>(agg, sElem);
  conv_layer<<<4096, 256, 0, stream>>>(s_h, beh, i1, i2, t_s2, b_sig2, t_o2, b_sof2, agg);
  residual_add<<<2048, 256, 0, stream>>>(s_f32, agg, s_h, sElem);

  // 5) pooling + head
  zero_f32<<<64, 256, 0, stream>>>(sums, (long)N_GRAPHS * EMB);
  zero_f32<<<2, 256, 0, stream>>>(cnts, (long)N_GRAPHS);
  pool_graphs<<<4096, 256, 0, stream>>>(s_f32, g2s, sums, cnts);
  head_mlp<<<N_GRAPHS, HH1, 0, stream>>>(sums, cnts, W1, b1, W2, b2, W3, b3, out);
}
